// DiscreteMamba2_10471130267688
// MI455X (gfx1250) — matrix operand order repaired and checked by reading, not compiled, not run on hardware
//
#include <hip/hip_runtime.h>
#include <cmath>

#define D_MODEL   1024
#define D_INNER   1024
#define NQK       32
#define NV        32
#define DSTATE    64
#define DCONV     4
#define CHUNK     128
#define CONV_DIM  5120          // D_INNER + 2*NQK*DSTATE
#define HEADDIM   32
#define IN_PROJ   6176          // 2*D_INNER + 2*NQK*DSTATE + NV
#define BSZ       2
#define LSEQ      8192
#define NC        64            // LSEQ / CHUNK
#define NTOK      16384         // BSZ * LSEQ

typedef __bf16 bf16;
typedef __bf16 bf16x16 __attribute__((ext_vector_type(16)));
typedef float  f32x8   __attribute__((ext_vector_type(8)));
typedef unsigned int u32x8 __attribute__((ext_vector_type(8)));
typedef unsigned int u32x4 __attribute__((ext_vector_type(4)));
typedef int i32x8v __attribute__((ext_vector_type(8)));
typedef int i32x4v __attribute__((ext_vector_type(4)));

union FragU { bf16x16 h; u32x8 u; };

__device__ inline f32x8 wmma_bf16(bf16x16 a, bf16x16 b, f32x8 c) {
  return __builtin_amdgcn_wmma_f32_16x16x32_bf16(
      false, a, false, b, (short)0, c, false, false);
}

// ---------------- Tensor Data Mover: 2D bf16 tile -> LDS ----------------
// D# per CDNA5 ISA. data_size=2B. pad_enable adds one
// DWORD of LDS padding per row (row = 2^(pad_code+1) DWORDs) so LDS row
// stride is (row_elems + 2) bf16 -> bank-conflict-free fragment reads.
__device__ inline void tdm_load_2d(unsigned lds_off, const bf16* gptr,
                                   unsigned td0, unsigned td1,
                                   unsigned tile0, unsigned tile1,
                                   unsigned long long stride_elems,
                                   unsigned pad_code) {
  unsigned long long ga = (unsigned long long)(size_t)gptr;
  u32x4 g0;
  g0[0] = 1u;                                         // count=1, user mode
  g0[1] = lds_off;                                    // LDS byte address
  g0[2] = (unsigned)(ga & 0xffffffffu);               // global_addr lo
  g0[3] = (unsigned)((ga >> 32) & 0x1ffffffu) | (2u << 30);  // ga hi | type=2
  i32x8v g1;
  g1[0] = (int)((1u << 16) | (1u << 20) | (pad_code << 22)); // 2B, pad_en, pad_amt=1dw
  g1[1] = (int)((td0 & 0xffffu) << 16);               // tensor_dim0 lo16 @ [63:48]
  g1[2] = (int)(((td0 >> 16) & 0xffffu) | ((td1 & 0xffffu) << 16));
  g1[3] = (int)(((td1 >> 16) & 0xffffu) | ((tile0 & 0xffffu) << 16));
  g1[4] = (int)(tile1 & 0xffffu);                     // tile_dim1, tile_dim2=0
  g1[5] = (int)(stride_elems & 0xffffffffull);        // dim0_stride lo
  g1[6] = (int)((stride_elems >> 32) & 0xffffull);    // dim0_stride hi
  g1[7] = 0;
  i32x4v z4 = {0, 0, 0, 0};
#if __has_include(<hip/amd_detail/amd_gfx1250_TDM.h>)
  i32x8v z8 = {0, 0, 0, 0, 0, 0, 0, 0};
  __builtin_amdgcn_tensor_load_to_lds(g0, g1, z4, z4, z8, 0);
#else
  __builtin_amdgcn_tensor_load_to_lds(g0, g1, z4, z4, 0);
#endif
}

__device__ inline unsigned lds_off_of(const void* p) {
  return (unsigned)(size_t)p;   // low 32 bits of flat shared address = LDS offset
}

// ---- fragment loaders (wave32 CDNA5 WMMA VGPR layouts) -----------------
__device__ inline bf16x16 load_a_frag(const bf16* base, int ld, int m0, int kk) {
  int lane = threadIdx.x & 31;
  int hl = lane >> 4;
  const bf16* row = base + (size_t)(m0 + (lane & 15)) * ld + kk + 16 * hl;
  FragU f;
#pragma unroll
  for (int v = 0; v < 8; ++v) {
    int k = 2 * v;
    f.u[v] = *(const unsigned int*)(row + k);
  }
  return f.h;
}

__device__ inline bf16x16 load_b_frag_nk(const bf16* base, int ld, int n0, int kk) {
  int lane = threadIdx.x & 31;
  int hl = lane >> 4;
  const bf16* row = base + (size_t)(n0 + (lane & 15)) * ld + kk + 16 * hl;
  FragU f;
#pragma unroll
  for (int v = 0; v < 8; ++v)
    f.u[v] = *(const unsigned int*)(row + 2 * v);
  return f.h;
}

__device__ inline bf16x16 load_b_frag_kn(const bf16* base, int ld, int n0, int kk) {
  int lane = threadIdx.x & 31;
  int hl = lane >> 4;
  int n = n0 + (lane & 15);
  bf16x16 r;
#pragma unroll
  for (int j = 0; j < 16; ++j) {
    int v = j >> 1, p = j & 1;
    int k = kk + 16 * hl + 2 * v + p;
    r[j] = base[(size_t)k * ld + n];
  }
  return r;
}

// =======================================================================
// GEMM: C[M,N] = A[M,K] * Bw[N,K]^T  (bf16 in, fp32 out)
// TDM double-buffered tiles; block 128x128; 8 waves 2(M)x4(N); K-step 32.
// =======================================================================
__global__ __launch_bounds__(256)
void gemm_bf16_kernel(const bf16* __restrict__ A, const bf16* __restrict__ Bw,
                      float* __restrict__ C, int M, int N, int K) {
  __shared__ bf16 As[2][128][34];
  __shared__ bf16 Bs[2][128][34];
  const int tid = threadIdx.x;
  const int lane = tid & 31, wave = tid >> 5;
  const int bm = blockIdx.y * 128, bn = blockIdx.x * 128;
  const int wm = (wave & 1) * 64, wn = (wave >> 1) * 32;
  const unsigned nrows = (N - bn < 128) ? (unsigned)(N - bn) : 128u;  // TDM zero-fills OOB rows
  f32x8 acc[4][2];
#pragma unroll
  for (int i = 0; i < 4; ++i)
#pragma unroll
    for (int j = 0; j < 2; ++j)
#pragma unroll
      for (int e = 0; e < 8; ++e) acc[i][j][e] = 0.f;

  if (wave == 0) {
    tdm_load_2d(lds_off_of(&As[0][0][0]), A + (size_t)bm * K, 32, 128, 32, 128, K, 3);
    tdm_load_2d(lds_off_of(&Bs[0][0][0]), Bw + (size_t)bn * K, 32, nrows, 32, 128, K, 3);
  }

  int buf = 0;
  for (int kk = 0; kk < K; kk += 32, buf ^= 1) {
    if (wave == 0) __builtin_amdgcn_s_wait_tensorcnt(0);
    __syncthreads();                         // current tiles ready for all waves
    if (wave == 0 && kk + 32 < K) {          // prefetch next K-slab via TDM
      tdm_load_2d(lds_off_of(&As[buf ^ 1][0][0]), A + (size_t)bm * K + kk + 32,
                  32, 128, 32, 128, K, 3);
      tdm_load_2d(lds_off_of(&Bs[buf ^ 1][0][0]), Bw + (size_t)bn * K + kk + 32,
                  32, nrows, 32, 128, K, 3);
    }
    bf16x16 afr[4], bfr[2];
#pragma unroll
    for (int mt = 0; mt < 4; ++mt) afr[mt] = load_a_frag(&As[buf][0][0], 34, wm + 16 * mt, 0);
#pragma unroll
    for (int nt = 0; nt < 2; ++nt) bfr[nt] = load_b_frag_nk(&Bs[buf][0][0], 34, wn + 16 * nt, 0);
#pragma unroll
    for (int mt = 0; mt < 4; ++mt)
#pragma unroll
      for (int nt = 0; nt < 2; ++nt)
        acc[mt][nt] = wmma_bf16(afr[mt], bfr[nt], acc[mt][nt]);
    __syncthreads();                         // done reading buf before TDM reuses it
  }

  int hl = lane >> 4;
#pragma unroll
  for (int mt = 0; mt < 4; ++mt)
#pragma unroll
    for (int nt = 0; nt < 2; ++nt)
#pragma unroll
      for (int r = 0; r < 8; ++r) {
        int gm = bm + wm + 16 * mt + r + 8 * hl;
        int gn = bn + wn + 16 * nt + (lane & 15);
        if (gn < N) C[(size_t)gm * N + gn] = acc[mt][nt][r];
      }
}

// ---------------- elementwise fp32 -> bf16 ------------------------------
__global__ __launch_bounds__(256)
void cvt_kernel(const float* __restrict__ src, bf16* __restrict__ dst, long long n) {
  long long gid = (long long)blockIdx.x * 256 + threadIdx.x;
  if (gid < n) dst[gid] = (bf16)src[gid];
}

// gated conversion for out_proj A-operand: yg = bf16(y * silu(z + zb))
__global__ __launch_bounds__(256)
void gate_kernel(const float* __restrict__ y, const float* __restrict__ xbcza,
                 const float* __restrict__ zb, bf16* __restrict__ yg) {
  long long gid = (long long)blockIdx.x * 256 + threadIdx.x;
  if (gid >= (long long)NTOK * D_INNER) return;
  int m = (int)(gid >> 10);
  int k = (int)(gid & 1023);
  float zv = xbcza[(size_t)m * IN_PROJ + CONV_DIM + k] + zb[k];
  yg[gid] = (bf16)(y[gid] * zv / (1.f + __expf(-zv)));
}

// ---------------- depthwise causal conv (fp32 in, bf16 out) -------------
__global__ __launch_bounds__(256)
void conv_kernel(const float* __restrict__ xbcza, const float* __restrict__ cw,
                 const float* __restrict__ cb, bf16* __restrict__ xconv) {
  long long gid = (long long)blockIdx.x * 256 + threadIdx.x;
  const long long total = (long long)NTOK * CONV_DIM;
  if (gid >= total) return;
  int c = (int)(gid % CONV_DIM);
  long long t = gid / CONV_DIM;
  int l = (int)(t % LSEQ);
  float acc = cb[c];
#pragma unroll
  for (int k = 0; k < DCONV; ++k) {
    int ll = l - (DCONV - 1) + k;
    if (ll >= 0)
      acc += cw[c * DCONV + k] * xbcza[(t - (DCONV - 1) + k) * IN_PROJ + c];
  }
  xconv[t * CONV_DIM + c] = (bf16)acc;
}

// dt = softplus(A_log)
__global__ __launch_bounds__(256)
void dt_kernel(const float* __restrict__ xbcza, float* __restrict__ dtp) {
  int gid = blockIdx.x * 256 + threadIdx.x;
  if (gid >= NTOK * NV) return;
  int h = gid & 31;
  int t = gid >> 5;
  float x = xbcza[(size_t)t * IN_PROJ + CONV_DIM + D_INNER + h];
  dtp[gid] = (x > 20.f) ? x : log1pf(__expf(x));
}

// =======================================================================
// Per (b, chunk, head): Y_diag + D*x, chunk states, chunk decay
// B/C tiles arrive via TDM; X via per-thread loads (needs fp32 copy too).
// =======================================================================
__global__ __launch_bounds__(256)
void chunk_kernel(const bf16* __restrict__ xconv, const float* __restrict__ dtp,
                  const float* __restrict__ Dv, float* __restrict__ y,
                  float* __restrict__ states, float* __restrict__ dc) {
  __shared__ bf16 Cb[128][66];
  __shared__ bf16 Bb[128][66];
  __shared__ float Xf[128][32];
  __shared__ bf16 Xb[128][34];
  __shared__ bf16 Xt[32][130];
  __shared__ bf16 Sb[128][130];
  __shared__ float Ac[128];
  int h = blockIdx.x & 31;
  int c = (blockIdx.x >> 5) & 63;
  int b = blockIdx.x >> 11;
  int tid = threadIdx.x, lane = tid & 31, wave = tid >> 5;
  size_t tok0 = (size_t)b * LSEQ + (size_t)c * CHUNK;

  if (wave == 0) {   // row = 64 bf16 = 32 DWORDs -> pad_code 4 -> LDS stride 66
    tdm_load_2d(lds_off_of(&Bb[0][0]),
                xconv + tok0 * CONV_DIM + D_INNER + h * DSTATE,
                64, 128, 64, 128, CONV_DIM, 4);
    tdm_load_2d(lds_off_of(&Cb[0][0]),
                xconv + tok0 * CONV_DIM + D_INNER + NQK * DSTATE + h * DSTATE,
                64, 128, 64, 128, CONV_DIM, 4);
  }
  if (tid < 128) Ac[tid] = dtp[(tok0 + tid) * NV + h];
#pragma unroll
  for (int t = 0; t < 16; ++t) {
    int idx = tid + t * 256;            // 4096 = 128*32
    int i = idx >> 5, p = idx & 31;
    bf16 v = xconv[(tok0 + i) * CONV_DIM + h * HEADDIM + p];
    Xf[i][p] = (float)v;
    Xb[i][p] = v;
  }
  if (wave == 0) __builtin_amdgcn_s_wait_tensorcnt(0);
  __syncthreads();
  if (tid == 0) {                        // Ac <- -cumsum(dt)
    float s = 0.f;
    for (int i = 0; i < 128; ++i) { s -= Ac[i]; Ac[i] = s; }
  }
  __syncthreads();
  float alast = Ac[127];
#pragma unroll
  for (int t = 0; t < 16; ++t) {         // Xt[p][l] = decay_states * X^T
    int idx = tid + t * 256;
    int p = idx >> 7, l = idx & 127;
    Xt[p][l] = (bf16)(__expf(alast - Ac[l]) * Xf[l][p]);
  }
  if (tid == 0) dc[((size_t)b * NV + h) * NC + c] = __expf(alast);
  __syncthreads();

  // ---- S = C * B^T : 16x128 strip per wave, K=64 ----------------------
  int m0 = 16 * wave;
  int hl = lane >> 4;
  f32x8 s[8];
#pragma unroll
  for (int nt = 0; nt < 8; ++nt)
#pragma unroll
    for (int e = 0; e < 8; ++e) s[nt][e] = 0.f;
#pragma unroll
  for (int ks = 0; ks < 2; ++ks) {
    bf16x16 a = load_a_frag(&Cb[0][0], 66, m0, 32 * ks);
#pragma unroll
    for (int nt = 0; nt < 8; ++nt) {
      bf16x16 bb = load_b_frag_nk(&Bb[0][0], 66, 16 * nt, 32 * ks);
      s[nt] = wmma_bf16(a, bb, s[nt]);
    }
  }
#pragma unroll
  for (int nt = 0; nt < 8; ++nt)
#pragma unroll
    for (int r = 0; r < 8; ++r) {
      int i = m0 + r + 8 * hl;
      int j = 16 * nt + (lane & 15);
      float v = s[nt][r];
      v = (i >= j) ? v * __expf(Ac[i] - Ac[j]) : 0.f;
      Sb[i][j] = (bf16)v;
    }
  __syncthreads();

  // ---- Y_diag = Sb * X : 16x32 per wave, K=128 ------------------------
  f32x8 yd[2];
#pragma unroll
  for (int nt = 0; nt < 2; ++nt)
#pragma unroll
    for (int e = 0; e < 8; ++e) yd[nt][e] = 0.f;
#pragma unroll
  for (int ks = 0; ks < 4; ++ks) {
    bf16x16 a = load_a_frag(&Sb[0][0], 130, m0, 32 * ks);
#pragma unroll
    for (int nt = 0; nt < 2; ++nt) {
      bf16x16 bb = load_b_frag_kn(&Xb[0][0], 34, 16 * nt, 32 * ks);
      yd[nt] = wmma_bf16(a, bb, yd[nt]);
    }
  }
  float Dh = Dv[h];
#pragma unroll
  for (int nt = 0; nt < 2; ++nt)
#pragma unroll
    for (int r = 0; r < 8; ++r) {
      int i = m0 + r + 8 * hl;
      int p = 16 * nt + (lane & 15);
      y[(tok0 + i) * D_INNER + h * HEADDIM + p] = yd[nt][r] + Dh * Xf[i][p];
    }

  // ---- states = Xt * B : one 16x16 tile per wave, K=128 ---------------
  int mt = wave & 1, ntw = wave >> 1;
  f32x8 st;
#pragma unroll
  for (int e = 0; e < 8; ++e) st[e] = 0.f;
#pragma unroll
  for (int ks = 0; ks < 4; ++ks) {
    bf16x16 a = load_a_frag(&Xt[0][0], 130, 16 * mt, 32 * ks);
    bf16x16 bb = load_b_frag_kn(&Bb[0][0], 66, 16 * ntw, 32 * ks);
    st = wmma_bf16(a, bb, st);
  }
  size_t sbase = (((size_t)b * NC + c) * NV + h) * (HEADDIM * DSTATE);
#pragma unroll
  for (int r = 0; r < 8; ++r) {
    int p = 16 * mt + r + 8 * hl;
    int n = 16 * ntw + (lane & 15);
    states[sbase + p * DSTATE + n] = st[r];
  }
}

// ---------------- inter-chunk scan --------------------------------------
__global__ __launch_bounds__(256)
void scan_kernel(const float* __restrict__ states, const float* __restrict__ dc,
                 float* __restrict__ prevs) {
  int b = blockIdx.x >> 5;
  int h = blockIdx.x & 31;
  int tid = threadIdx.x;
  float S[8];
#pragma unroll
  for (int j = 0; j < 8; ++j) S[j] = 0.f;
  for (int c = 0; c < NC; ++c) {
    float d = dc[((size_t)b * NV + h) * NC + c];
    size_t base = (((size_t)b * NC + c) * NV + h) * (HEADDIM * DSTATE);
#pragma unroll
    for (int j = 0; j < 8; ++j) {
      int idx = tid + j * 256;
      prevs[base + idx] = S[j];
      S[j] = S[j] * d + states[base + idx];
    }
  }
}

// ---------------- Y_off = exp(Acum) ∘ (C * prevs^T), += into y ----------
__global__ __launch_bounds__(256)
void yoff_kernel(const bf16* __restrict__ xconv, const float* __restrict__ dtp,
                 const float* __restrict__ prevs, float* __restrict__ y) {
  __shared__ bf16 Cb[128][66];
  __shared__ bf16 Pb[32][66];
  __shared__ float Ac[128];
  int h = blockIdx.x & 31;
  int c = (blockIdx.x >> 5) & 63;
  int b = blockIdx.x >> 11;
  int tid = threadIdx.x, lane = tid & 31, wave = tid >> 5;
  size_t tok0 = (size_t)b * LSEQ + (size_t)c * CHUNK;

  if (wave == 0)
    tdm_load_2d(lds_off_of(&Cb[0][0]),
                xconv + tok0 * CONV_DIM + D_INNER + NQK * DSTATE + h * DSTATE,
                64, 128, 64, 128, CONV_DIM, 4);
  if (tid < 128) Ac[tid] = dtp[(tok0 + tid) * NV + h];
  size_t pbase = (((size_t)b * NC + c) * NV + h) * (HEADDIM * DSTATE);
#pragma unroll
  for (int t = 0; t < 8; ++t) {
    int idx = tid + t * 256;            // 2048 = 32*64
    int p = idx >> 6, n = idx & 63;
    Pb[p][n] = (bf16)prevs[pbase + p * DSTATE + n];
  }
  if (wave == 0) __builtin_amdgcn_s_wait_tensorcnt(0);
  __syncthreads();
  if (tid == 0) {
    float s = 0.f;
    for (int i = 0; i < 128; ++i) { s -= Ac[i]; Ac[i] = s; }
  }
  __syncthreads();

  int m0 = 16 * wave, hl = lane >> 4;
  f32x8 acc[2];
#pragma unroll
  for (int nt = 0; nt < 2; ++nt)
#pragma unroll
    for (int e = 0; e < 8; ++e) acc[nt][e] = 0.f;
#pragma unroll
  for (int ks = 0; ks < 2; ++ks) {
    bf16x16 a = load_a_frag(&Cb[0][0], 66, m0, 32 * ks);
#pragma unroll
    for (int nt = 0; nt < 2; ++nt) {
      bf16x16 bb = load_b_frag_nk(&Pb[0][0], 66, 16 * nt, 32 * ks);
      acc[nt] = wmma_bf16(a, bb, acc[nt]);
    }
  }
#pragma unroll
  for (int nt = 0; nt < 2; ++nt)
#pragma unroll
    for (int r = 0; r < 8; ++r) {
      int i = m0 + r + 8 * hl;
      int p = 16 * nt + (lane & 15);
      size_t oidx = (tok0 + i) * D_INNER + h * HEADDIM + p;
      y[oidx] += __expf(Ac[i]) * acc[nt][r];
    }
}

// =======================================================================
extern "C" void kernel_launch(void* const* d_in, const int* in_sizes, int n_in,
                              void* d_out, int out_size, void* d_ws, size_t ws_size,
                              hipStream_t stream) {
  const float* u      = (const float*)d_in[0];
  const float* W_in   = (const float*)d_in[1];
  const float* conv_w = (const float*)d_in[2];
  const float* conv_b = (const float*)d_in[3];
  const float* z_bias = (const float*)d_in[4];
  const float* Dv     = (const float*)d_in[5];
  const float* W_out  = (const float*)d_in[6];
  float* out = (float*)d_out;
  char* wsb = (char*)d_ws;

  size_t off = 0;
  auto alloc_f = [&](size_t n) { float* p = (float*)(wsb + off); off += n * 4; return p; };
  auto alloc_h = [&](size_t n) { bf16* p = (bf16*)(wsb + off); off += ((n * 2 + 255) & ~(size_t)255); return p; };

  float* xbcza  = alloc_f((size_t)NTOK * IN_PROJ);
  float* yb     = alloc_f((size_t)NTOK * D_INNER);
  float* dtb    = alloc_f((size_t)NTOK * NV);
  float* states = alloc_f((size_t)BSZ * NC * NV * HEADDIM * DSTATE);
  float* prevs  = alloc_f((size_t)BSZ * NC * NV * HEADDIM * DSTATE);
  float* dcb    = alloc_f((size_t)BSZ * NV * NC);
  bf16* u_h     = alloc_h((size_t)NTOK * D_MODEL);
  bf16* win_h   = alloc_h((size_t)IN_PROJ * D_MODEL);
  bf16* wout_h  = alloc_h((size_t)D_MODEL * D_INNER);
  bf16* xconv   = alloc_h((size_t)NTOK * CONV_DIM);
  bf16* yg_h    = alloc_h((size_t)NTOK * D_INNER);

  // 0) one-shot bf16 conversions (operands then live in 192MB L2)
  long long nu = (long long)NTOK * D_MODEL;
  cvt_kernel<<<(unsigned)((nu + 255) / 256), 256, 0, stream>>>(u, u_h, nu);
  long long nwi = (long long)IN_PROJ * D_MODEL;
  cvt_kernel<<<(unsigned)((nwi + 255) / 256), 256, 0, stream>>>(W_in, win_h, nwi);
  long long nwo = (long long)D_MODEL * D_INNER;
  cvt_kernel<<<(unsigned)((nwo + 255) / 256), 256, 0, stream>>>(W_out, wout_h, nwo);

  // 1) in_proj GEMM: xBCzA = u @ W_in^T
  dim3 g1((IN_PROJ + 127) / 128, NTOK / 128);
  gemm_bf16_kernel<<<g1, 256, 0, stream>>>(u_h, win_h, xbcza, NTOK, IN_PROJ, D_MODEL);
  // 2) depthwise causal conv -> bf16
  long long ctotal = (long long)NTOK * CONV_DIM;
  conv_kernel<<<(unsigned)((ctotal + 255) / 256), 256, 0, stream>>>(xbcza, conv_w, conv_b, xconv);
  // 3) dt = softplus(A_log)
  dt_kernel<<<(NTOK * NV + 255) / 256, 256, 0, stream>>>(xbcza, dtb);
  // 4) intra-chunk WMMA
  chunk_kernel<<<BSZ * NC * NV, 256, 0, stream>>>(xconv, dtb, Dv, yb, states, dcb);
  // 5) inter-chunk scan
  scan_kernel<<<BSZ * NV, 256, 0, stream>>>(states, dcb, prevs);
  // 6) Y_off accumulation
  yoff_kernel<<<BSZ * NC * NV, 256, 0, stream>>>(xconv, dtb, prevs, yb);
  // 7) gate + out_proj GEMM
  long long ng = (long long)NTOK * D_INNER;
  gate_kernel<<<(unsigned)((ng + 255) / 256), 256, 0, stream>>>(yb, xbcza, z_bias, yg_h);
  dim3 g2(D_MODEL / 128, NTOK / 128);
  gemm_bf16_kernel<<<g2, 256, 0, stream>>>(yg_h, wout_h, out, NTOK, D_MODEL, D_INNER);
  (void)in_sizes; (void)n_in; (void)out_size; (void)ws_size;
}
